// FocusedDecoder_89713276879189
// MI455X (gfx1250) — matrix operand order repaired and checked by reading, not compiled, not run on hardware
//
#include <hip/hip_runtime.h>
#include <math.h>

#define BB 2
#define DDIM 256
#define HH 8
#define HD 32
#define FF 1024
#define LLAY 6
#define NQ 540
#define PTOT 12800

typedef __attribute__((ext_vector_type(16))) _Float16 v16h;
typedef __attribute__((ext_vector_type(8)))  float    v8f;

// ---------------------------------------------------------------------------
// WMMA fragment loaders (wave32; this code uses contiguous k = 16*(l/16) + e in both A and B)
// A-matrix 16x32 f16: lane l -> M = l&15; halves e: K = e + (l>=16?16:0)
__device__ __forceinline__ v16h frag_a_lds(const _Float16* s, int ld) {
  const int l = threadIdx.x & 31;
  const int m = l & 15;
  const int kh = (l >> 4) << 4;
  v16h r;
#pragma unroll
  for (int e = 0; e < 16; ++e) {
    int k = e + kh;
    r[e] = s[m * ld + k];
  }
  return r;
}
// B-matrix 32x16 f16, Bmat[k][n] = s[n*ld + k]  (n-major storage, e.g. weight rows)
__device__ __forceinline__ v16h frag_b_nmajor(const _Float16* s, int ld) {
  const int l = threadIdx.x & 31;
  const int n = l & 15;
  const int kh = (l >> 4) << 4;
  v16h r;
#pragma unroll
  for (int e = 0; e < 16; ++e) r[e] = s[n * ld + kh + e];
  return r;
}
// B-matrix 32x16 f16, Bmat[k][n] = s[k*ld + n]  (k-major storage, e.g. V tiles)
__device__ __forceinline__ v16h frag_b_kmajor(const _Float16* s, int ld) {
  const int l = threadIdx.x & 31;
  const int n = l & 15;
  const int kh = (l >> 4) << 4;
  v16h r;
#pragma unroll
  for (int e = 0; e < 16; ++e) r[e] = s[(kh + e) * ld + n];
  return r;
}

#define WMMA_F16(A, B, C) \
  __builtin_amdgcn_wmma_f32_16x16x32_f16(false, (A), false, (B), (short)0, (C), false, false)

// Stage 16 consecutive f32 -> f16 into LDS (16B-aligned source guaranteed by callers)
__device__ __forceinline__ void stage16(_Float16* dst, const float* src, bool valid) {
  float t[16];
  if (valid) {
    const float4* p = reinterpret_cast<const float4*>(src);
#pragma unroll
    for (int i = 0; i < 4; ++i) {
      float4 f = p[i];
      t[4 * i + 0] = f.x; t[4 * i + 1] = f.y; t[4 * i + 2] = f.z; t[4 * i + 3] = f.w;
    }
  } else {
#pragma unroll
    for (int i = 0; i < 16; ++i) t[i] = 0.f;
  }
#pragma unroll
  for (int i = 0; i < 16; ++i) dst[i] = (_Float16)t[i];
}

// ---------------------------------------------------------------------------
// Async global->LDS copy of 16 bytes per lane (GLOBAL_LOAD_ASYNC_TO_LDS_B128,
// tracked by ASYNCcnt). Generic LDS pointers are {aperture_hi, lds_offset}, so
// the low 32 bits of a __shared__ pointer are the wave-relative LDS address.
__device__ __forceinline__ uint32_t lds_offset(const void* p) {
  return (uint32_t)(uintptr_t)p;
}
__device__ __forceinline__ void async_ld_b128(uint32_t lds_byte_off, const void* g) {
  asm volatile("global_load_async_to_lds_b128 %0, %1, off"
               :: "v"(lds_byte_off), "v"((unsigned long long)(uintptr_t)g)
               : "memory");
}
__device__ __forceinline__ void wait_asynccnt0() {
  asm volatile("s_wait_asynccnt 0x0" ::: "memory");
}

// ---------------------------------------------------------------------------
// Generic GEMM: C[m,n] = relu?(( sum_k A[m,k]*W[n,k] + bias[n]) * alpha)
// A: (M,K) f32 row-major, W: (N,K) f32 row-major. 64x64 tile, 128 threads.
// out_f16: write C as f16 (for attention operand buffers), else f32.
__global__ void __launch_bounds__(128)
gemm_xwT(const float* __restrict__ A, const float* __restrict__ W,
         const float* __restrict__ bias, void* __restrict__ Cv,
         int M, int N, int K, float alpha, int relu, int out_f16) {
  __shared__ _Float16 As[64 * 40];
  __shared__ _Float16 Ws[64 * 40];
  const int tid = threadIdx.x;
  const int wave = tid >> 5, lane = tid & 31;
  const int mh = wave >> 1, nh = wave & 1;
  const int m0 = blockIdx.y * 64;
  const int n0 = blockIdx.x * 64;

  v8f acc[2][2];
#pragma unroll
  for (int mi = 0; mi < 2; ++mi)
#pragma unroll
    for (int ni = 0; ni < 2; ++ni)
      acc[mi][ni] = (v8f){0.f, 0.f, 0.f, 0.f, 0.f, 0.f, 0.f, 0.f};

  const int srow = tid >> 1;
  const int skb = (tid & 1) << 4;

  for (int k0 = 0; k0 < K; k0 += 32) {
    const int gm = m0 + srow;
    stage16(As + srow * 40 + skb, A + (size_t)gm * K + k0 + skb, gm < M);
    const int gn = n0 + srow;
    stage16(Ws + srow * 40 + skb, W + (size_t)gn * K + k0 + skb, gn < N);
    __syncthreads();

    v16h af[2], bf[2];
#pragma unroll
    for (int mi = 0; mi < 2; ++mi) af[mi] = frag_a_lds(As + (mh * 32 + mi * 16) * 40, 40);
#pragma unroll
    for (int ni = 0; ni < 2; ++ni) bf[ni] = frag_b_nmajor(Ws + (nh * 32 + ni * 16) * 40, 40);
#pragma unroll
    for (int mi = 0; mi < 2; ++mi)
#pragma unroll
      for (int ni = 0; ni < 2; ++ni)
        acc[mi][ni] = WMMA_F16(af[mi], bf[ni], acc[mi][ni]);
    __syncthreads();
  }

  float* Cf = (float*)Cv;
  _Float16* Ch = (_Float16*)Cv;
#pragma unroll
  for (int mi = 0; mi < 2; ++mi)
#pragma unroll
    for (int ni = 0; ni < 2; ++ni) {
      const int ncol = n0 + nh * 32 + ni * 16 + (lane & 15);
      const float bv = (bias != nullptr && ncol < N) ? bias[ncol] : 0.f;
#pragma unroll
      for (int r = 0; r < 8; ++r) {
        const int mrow = m0 + mh * 32 + mi * 16 + r + ((lane >> 4) << 3);
        if (mrow < M && ncol < N) {
          float v = (acc[mi][ni][r] + bv) * alpha;
          if (relu) v = fmaxf(v, 0.f);
          if (out_f16) Ch[(size_t)mrow * N + ncol] = (_Float16)v;
          else         Cf[(size_t)mrow * N + ncol] = v;
        }
      }
    }
}

// ---------------------------------------------------------------------------
// Flash attention: per (b, h, 64-query tile). Q/K/V are (B, L, 256) f16 with
// head offset h*32; O is (B, Lq, 256) f32. Q is pre-scaled by 1/sqrt(HD).
// mask_mode: 0 = none (only k < Lk), 1 = spatial box mask from
// bnds[(b*Lq+q)*6] = (lo0,lo1,lo2,hi0,hi1,hi2). K/V chunks are streamed into
// LDS with GLOBAL_LOAD_ASYNC_TO_LDS_B128.
__global__ void __launch_bounds__(128)
flash_attn(const _Float16* __restrict__ Q, const _Float16* __restrict__ Kb,
           const _Float16* __restrict__ Vb, float* __restrict__ O,
           const int* __restrict__ bnds, int Lq, int Lk, int mask_mode) {
  __shared__ __attribute__((aligned(16))) _Float16 Qs[64 * 40];
  __shared__ __attribute__((aligned(16))) _Float16 Ks[64 * 40];
  __shared__ __attribute__((aligned(16))) _Float16 Vs[64 * 40];
  __shared__ __attribute__((aligned(16))) _Float16 Ps[64 * 72];
  __shared__ float s_pmax[64][2];
  __shared__ float s_psum[64][2];
  __shared__ float s_runmax[64];
  __shared__ float s_runsum[64];
  __shared__ float s_resc[64];
  __shared__ float s_nmax[64];
  __shared__ int   s_lo[64][3];
  __shared__ int   s_hi[64][3];

  const int tid = threadIdx.x;
  const int wave = tid >> 5, lane = tid & 31;
  const int mh = wave >> 1;
  const int nh = wave & 1;   // role in S stage
  const int hdt = wave & 1;  // role in O stage
  const int q0 = blockIdx.x * 64;
  const int h = blockIdx.y;
  const int b = blockIdx.z;

  const int sr = tid >> 1;            // staging row (64 rows, 2 threads each)
  const int skb = (tid & 1) << 4;     // staging half-offset (0 or 16 halves)
  const uint32_t ksoff = lds_offset(Ks) + (uint32_t)(sr * 40 + skb) * 2;
  const uint32_t vsoff = lds_offset(Vs) + (uint32_t)(sr * 40 + skb) * 2;

  // stage Q tile (64 x 32 halves)
  {
    const int gq = q0 + sr;
    uint32_t* d = (uint32_t*)(Qs + sr * 40 + skb);
    if (gq < Lq) {
      const uint32_t* p =
          (const uint32_t*)(Q + ((size_t)(b * Lq + gq)) * DDIM + h * HD + skb);
#pragma unroll
      for (int i = 0; i < 8; ++i) d[i] = p[i];
    } else {
#pragma unroll
      for (int i = 0; i < 8; ++i) d[i] = 0u;
    }
  }
  if (tid < 64) {
    s_runmax[tid] = -1e30f;
    s_runsum[tid] = 0.f;
    if (mask_mode) {
      int gq = q0 + tid; if (gq >= Lq) gq = Lq - 1;
      const int* bp = bnds + ((size_t)(b * Lq + gq)) * 6;
#pragma unroll
      for (int j = 0; j < 3; ++j) { s_lo[tid][j] = bp[j]; s_hi[tid][j] = bp[3 + j]; }
    }
  }
  v8f oacc[2];
  oacc[0] = (v8f){0.f, 0.f, 0.f, 0.f, 0.f, 0.f, 0.f, 0.f};
  oacc[1] = oacc[0];

  for (int p0 = 0; p0 < Lk; p0 += 64) {
    __syncthreads();  // protect Ks/Vs/Ps from previous iteration readers
    {
      const int gp = p0 + sr;
      if (gp < Lk) {
        const _Float16* kr = Kb + ((size_t)(b * Lk + gp)) * DDIM + h * HD + skb;
        const _Float16* vr = Vb + ((size_t)(b * Lk + gp)) * DDIM + h * HD + skb;
        async_ld_b128(ksoff,      kr);
        async_ld_b128(ksoff + 16, kr + 8);
        async_ld_b128(vsoff,      vr);
        async_ld_b128(vsoff + 16, vr + 8);
      } else {
        uint32_t* dk = (uint32_t*)(Ks + sr * 40 + skb);
        uint32_t* dv = (uint32_t*)(Vs + sr * 40 + skb);
#pragma unroll
        for (int i = 0; i < 8; ++i) { dk[i] = 0u; dv[i] = 0u; }
      }
    }
    wait_asynccnt0();
    __syncthreads();

    // S = Q Kchunk^T (each wave: 32x32 quadrant, 2x2 wmma tiles)
    v8f sacc[2][2];
#pragma unroll
    for (int mi = 0; mi < 2; ++mi)
#pragma unroll
      for (int ni = 0; ni < 2; ++ni)
        sacc[mi][ni] = (v8f){0.f, 0.f, 0.f, 0.f, 0.f, 0.f, 0.f, 0.f};
    {
      v16h af[2], bf[2];
#pragma unroll
      for (int mi = 0; mi < 2; ++mi) af[mi] = frag_a_lds(Qs + (mh * 32 + mi * 16) * 40, 40);
#pragma unroll
      for (int ni = 0; ni < 2; ++ni) bf[ni] = frag_b_nmajor(Ks + (nh * 32 + ni * 16) * 40, 40);
#pragma unroll
      for (int mi = 0; mi < 2; ++mi)
#pragma unroll
        for (int ni = 0; ni < 2; ++ni)
          sacc[mi][ni] = WMMA_F16(af[mi], bf[ni], sacc[mi][ni]);
    }
    // mask
#pragma unroll
    for (int mi = 0; mi < 2; ++mi)
#pragma unroll
      for (int ni = 0; ni < 2; ++ni)
#pragma unroll
        for (int r = 0; r < 8; ++r) {
          const int lrow = mh * 32 + mi * 16 + r + ((lane >> 4) << 3);
          const int p = p0 + nh * 32 + ni * 16 + (lane & 15);
          bool valid = p < Lk;
          if (valid && mask_mode) {
            const int x = p / 640, y = (p >> 5) % 20, z = p & 31;
            valid = x >= s_lo[lrow][0] && x < s_hi[lrow][0] &&
                    y >= s_lo[lrow][1] && y < s_hi[lrow][1] &&
                    z >= s_lo[lrow][2] && z < s_hi[lrow][2];
          }
          if (!valid) sacc[mi][ni][r] = -1e30f;
        }
    // per-row max over this wave's 32 columns (16-lane xor reduction per half)
#pragma unroll
    for (int mi = 0; mi < 2; ++mi)
#pragma unroll
      for (int r = 0; r < 8; ++r) {
        float v = fmaxf(sacc[mi][0][r], sacc[mi][1][r]);
#pragma unroll
        for (int off = 1; off < 16; off <<= 1) v = fmaxf(v, __shfl_xor(v, off, 32));
        const int lrow = mh * 32 + mi * 16 + r + ((lane >> 4) << 3);
        if (lane == 0 || lane == 16) s_pmax[lrow][nh] = v;
      }
    __syncthreads();
    if (tid < 64) {
      const float cm = fmaxf(s_pmax[tid][0], s_pmax[tid][1]);
      const float om = s_runmax[tid];
      const float nm = fmaxf(om, cm);
      s_nmax[tid] = nm;
      s_resc[tid] = __expf(om - nm);
      s_runmax[tid] = nm;
    }
    __syncthreads();
    // P = exp(S - newmax), write f16 tile + partial row sums
#pragma unroll
    for (int mi = 0; mi < 2; ++mi)
#pragma unroll
      for (int ni = 0; ni < 2; ++ni)
#pragma unroll
        for (int r = 0; r < 8; ++r) {
          const int lrow = mh * 32 + mi * 16 + r + ((lane >> 4) << 3);
          const float s = sacc[mi][ni][r];
          const float pv = (s > -1e29f) ? __expf(s - s_nmax[lrow]) : 0.f;
          sacc[mi][ni][r] = pv;
          Ps[lrow * 72 + nh * 32 + ni * 16 + (lane & 15)] = (_Float16)pv;
        }
#pragma unroll
    for (int mi = 0; mi < 2; ++mi)
#pragma unroll
      for (int r = 0; r < 8; ++r) {
        float v = sacc[mi][0][r] + sacc[mi][1][r];
#pragma unroll
        for (int off = 1; off < 16; off <<= 1) v += __shfl_xor(v, off, 32);
        const int lrow = mh * 32 + mi * 16 + r + ((lane >> 4) << 3);
        if (lane == 0 || lane == 16) s_psum[lrow][nh] = v;
      }
    __syncthreads();
    if (tid < 64)
      s_runsum[tid] = s_runsum[tid] * s_resc[tid] + s_psum[tid][0] + s_psum[tid][1];
    // O stage: rescale accumulators, then O += P @ Vchunk (wave = (mh, hdt))
#pragma unroll
    for (int mi = 0; mi < 2; ++mi)
#pragma unroll
      for (int r = 0; r < 8; ++r) {
        const int lrow = mh * 32 + mi * 16 + r + ((lane >> 4) << 3);
        oacc[mi][r] *= s_resc[lrow];
      }
#pragma unroll
    for (int ks = 0; ks < 2; ++ks) {
      v16h bf = frag_b_kmajor(Vs + (ks * 32) * 40 + hdt * 16, 40);
#pragma unroll
      for (int mi = 0; mi < 2; ++mi) {
        v16h af = frag_a_lds(Ps + (mh * 32 + mi * 16) * 72 + ks * 32, 72);
        oacc[mi] = WMMA_F16(af, bf, oacc[mi]);
      }
    }
  }
  __syncthreads();
#pragma unroll
  for (int mi = 0; mi < 2; ++mi)
#pragma unroll
    for (int r = 0; r < 8; ++r) {
      const int lrow = mh * 32 + mi * 16 + r + ((lane >> 4) << 3);
      const int gq = q0 + lrow;
      if (gq < Lq) {
        const float v = oacc[mi][r] / s_runsum[lrow];
        O[((size_t)(b * Lq + gq)) * DDIM + h * HD + hdt * 16 + (lane & 15)] = v;
      }
    }
}

// ---------------------------------------------------------------------------
__global__ void prep_kernel(const float* __restrict__ src, const float* __restrict__ pos,
                            float* __restrict__ kvT, float* __restrict__ srcT) {
  size_t i = (size_t)blockIdx.x * 256 + threadIdx.x;
  if (i >= (size_t)BB * DDIM * PTOT) return;
  const int p = (int)(i % PTOT);
  const int d = (int)((i / PTOT) % DDIM);
  const int b = (int)(i / ((size_t)PTOT * DDIM));
  const float s = src[i];
  const float pp = pos[i];
  const size_t o = ((size_t)b * PTOT + p) * DDIM + d;
  srcT[o] = s;
  kvT[o] = s + pp;
}

__global__ void init_tgt_kernel(const float* __restrict__ qe, float* __restrict__ tgt) {
  size_t i = (size_t)blockIdx.x * 256 + threadIdx.x;
  if (i >= (size_t)BB * NQ * DDIM) return;
  const int d = (int)(i % DDIM);
  const int q = (int)((i / DDIM) % NQ);
  tgt[i] = qe[(size_t)q * 512 + 256 + d];
}

__global__ void add_qpos_kernel(const float* __restrict__ tgt, const float* __restrict__ qe,
                                float* __restrict__ out) {
  size_t i = (size_t)blockIdx.x * 256 + threadIdx.x;
  if (i >= (size_t)BB * NQ * DDIM) return;
  const int d = (int)(i % DDIM);
  const int q = (int)((i / DDIM) % NQ);
  out[i] = tgt[i] + qe[(size_t)q * 512 + d];
}

__global__ void __launch_bounds__(256)
ln_residual_kernel(float* __restrict__ x, const float* __restrict__ t2,
                   const float* __restrict__ g, const float* __restrict__ bt) {
  __shared__ float red[256];
  const int row = blockIdx.x, t = threadIdx.x;
  const size_t o = (size_t)row * DDIM + t;
  const float v = x[o] + t2[o];
  red[t] = v;
  __syncthreads();
  for (int s = 128; s > 0; s >>= 1) { if (t < s) red[t] += red[t + s]; __syncthreads(); }
  const float mean = red[0] / (float)DDIM;
  __syncthreads();
  const float d = v - mean;
  red[t] = d * d;
  __syncthreads();
  for (int s = 128; s > 0; s >>= 1) { if (t < s) red[t] += red[t + s]; __syncthreads(); }
  const float var = red[0] / (float)DDIM;
  x[o] = d * rsqrtf(var + 1e-5f) * g[t] + bt[t];
}

// bounds: lo/hi ints per (b,q). mode 0: from attn_area; mode 1: reg head on h2.
__global__ void bounds_kernel(const float* __restrict__ h2, const float* __restrict__ w3,
                              const float* __restrict__ b3, const float* __restrict__ anchors,
                              const float* __restrict__ attn_area, int* __restrict__ bnds,
                              int mode, float pad0, float pad1, float pad2) {
  const int idx = blockIdx.x * blockDim.x + threadIdx.x;
  if (idx >= BB * NQ) return;
  const int q = idx % NQ;
  float vols[6];
  if (mode == 0) {
    const int organ = q / 27;
#pragma unroll
    for (int j = 0; j < 6; ++j) vols[j] = attn_area[organ * 6 + j];
  } else {
    float prop[6];
#pragma unroll
    for (int j = 0; j < 6; ++j) {
      float s = b3[j];
      const float* hr = h2 + (size_t)idx * DDIM;
      const float* wr = w3 + (size_t)j * DDIM;
      for (int k = 0; k < DDIM; ++k) s += hr[k] * wr[k];
      float pv = tanhf(s) * 0.1f + anchors[q * 6 + j];
      prop[j] = fminf(fmaxf(pv, 0.f), 1.f);
    }
#pragma unroll
    for (int c = 0; c < 3; ++c) {
      vols[c] = prop[c] - 0.5f * prop[3 + c];
      vols[3 + c] = prop[c] + 0.5f * prop[3 + c];
    }
  }
  const float shp[3] = {20.f, 20.f, 32.f};
  const float pad[3] = {pad0, pad1, pad2};
#pragma unroll
  for (int d = 0; d < 3; ++d) {
    float lo = floorf(vols[d] * shp[d] - pad[d]);
    float hi = floorf(vols[3 + d] * shp[d] + pad[d]);
    lo = fminf(fmaxf(lo, 0.f), shp[d]);
    hi = fminf(fmaxf(hi, 0.f), shp[d]);
    bnds[(size_t)idx * 6 + d] = (int)lo;
    bnds[(size_t)idx * 6 + 3 + d] = (int)hi;
  }
}

__global__ void copy_kernel(const float* __restrict__ a, float* __restrict__ o, int n) {
  int i = blockIdx.x * 256 + threadIdx.x;
  if (i < n) o[i] = a[i];
}

// ---------------------------------------------------------------------------
extern "C" void kernel_launch(void* const* d_in, const int* in_sizes, int n_in,
                              void* d_out, int out_size, void* d_ws, size_t ws_size,
                              hipStream_t stream) {
  const float* src      = (const float*)d_in[0];
  const float* pos      = (const float*)d_in[1];
  const float* qe       = (const float*)d_in[2];
  const float* sa_in_w  = (const float*)d_in[3];
  const float* sa_in_b  = (const float*)d_in[4];
  const float* sa_out_w = (const float*)d_in[5];
  const float* sa_out_b = (const float*)d_in[6];
  const float* ca_k_w   = (const float*)d_in[7];
  const float* ca_v_w   = (const float*)d_in[8];
  const float* ca_p_w   = (const float*)d_in[9];
  const float* ca_p_b   = (const float*)d_in[10];
  const float* ln1_g    = (const float*)d_in[11];
  const float* ln1_b    = (const float*)d_in[12];
  const float* ln2_g    = (const float*)d_in[13];
  const float* ln2_b    = (const float*)d_in[14];
  const float* ln3_g    = (const float*)d_in[15];
  const float* ln3_b    = (const float*)d_in[16];
  const float* ffn_w1   = (const float*)d_in[17];
  const float* ffn_b1   = (const float*)d_in[18];
  const float* ffn_w2   = (const float*)d_in[19];
  const float* ffn_b2   = (const float*)d_in[20];
  const float* reg_w1   = (const float*)d_in[21];
  const float* reg_b1   = (const float*)d_in[22];
  const float* reg_w2   = (const float*)d_in[23];
  const float* reg_b2   = (const float*)d_in[24];
  const float* reg_w3   = (const float*)d_in[25];
  const float* reg_b3   = (const float*)d_in[26];
  const float* anchors  = (const float*)d_in[27];
  const float* attnarea = (const float*)d_in[28];

  float* ws = (float*)d_ws;
  size_t off = 0;
  auto alloc = [&](size_t nfloats) {
    float* p = ws + off;
    off += (nfloats + 63) & ~(size_t)63;
    return p;
  };
  const size_t MQ = (size_t)BB * NQ;          // 1080 rows
  const size_t MP = (size_t)BB * PTOT;        // 25600 rows
  float*    kvT  = alloc(MP * DDIM);
  float*    srcT = alloc(MP * DDIM);
  _Float16* Kc   = (_Float16*)alloc(MP * DDIM / 2);   // f16 K (cross)
  _Float16* Vc   = (_Float16*)alloc(MP * DDIM / 2);   // f16 V (cross)
  float*    tgt  = alloc(MQ * DDIM);
  float*    qk   = alloc(MQ * DDIM);
  _Float16* qb   = (_Float16*)alloc(MQ * DDIM / 2);   // f16 q (self)
  _Float16* kb   = (_Float16*)alloc(MQ * DDIM / 2);   // f16 k (self)
  _Float16* vb   = (_Float16*)alloc(MQ * DDIM / 2);   // f16 v (self)
  _Float16* qc   = (_Float16*)alloc(MQ * DDIM / 2);   // f16 q (cross)
  float*    ob   = alloc(MQ * DDIM);
  float*    t2   = alloc(MQ * DDIM);
  float*    h1   = alloc(MQ * FF);
  float*    h2   = alloc(MQ * DDIM);
  int*      bnds = (int*)alloc(MQ * 6);

  const float qscale = 0.17677669529663687f;  // 1/sqrt(32)
  const float FOCUS[LLAY] = {0.1f, 0.075f, 0.05f, 0.05f, 0.025f, 0.025f};

  {
    const size_t n = (size_t)BB * DDIM * PTOT;
    prep_kernel<<<(unsigned)((n + 255) / 256), 256, 0, stream>>>(src, pos, kvT, srcT);
  }
  init_tgt_kernel<<<(unsigned)((MQ * DDIM + 255) / 256), 256, 0, stream>>>(qe, tgt);

  const dim3 gQ(DDIM / 64, (unsigned)((MQ + 63) / 64));     // (4, 17)
  const dim3 gP(DDIM / 64, (unsigned)(MP / 64));            // (4, 400)
  const dim3 gF(FF / 64, (unsigned)((MQ + 63) / 64));       // (16, 17)
  const dim3 gEl((unsigned)((MQ * DDIM + 255) / 256));
  const dim3 gFlash((NQ + 63) / 64, HH, BB);                // (9, 8, 2)

  for (int i = 0; i < LLAY; ++i) {
    const float* wq = sa_in_w + (size_t)i * 3 * DDIM * DDIM;
    const float* wk = wq + (size_t)DDIM * DDIM;
    const float* wv = wq + (size_t)2 * DDIM * DDIM;
    const float* bq = sa_in_b + (size_t)i * 3 * DDIM;
    const float* bk = bq + DDIM;
    const float* bv = bq + 2 * DDIM;

    // ---- self attention ----
    add_qpos_kernel<<<gEl, 256, 0, stream>>>(tgt, qe, qk);
    gemm_xwT<<<gQ, 128, 0, stream>>>(qk, wq, bq, qb, (int)MQ, DDIM, DDIM, qscale, 0, 1);
    gemm_xwT<<<gQ, 128, 0, stream>>>(qk, wk, bk, kb, (int)MQ, DDIM, DDIM, 1.f, 0, 1);
    gemm_xwT<<<gQ, 128, 0, stream>>>(tgt, wv, bv, vb, (int)MQ, DDIM, DDIM, 1.f, 0, 1);
    flash_attn<<<gFlash, 128, 0, stream>>>(qb, kb, vb, ob, bnds, NQ, NQ, 0);
    gemm_xwT<<<gQ, 128, 0, stream>>>(ob, sa_out_w + (size_t)i * DDIM * DDIM,
                                     sa_out_b + (size_t)i * DDIM, t2,
                                     (int)MQ, DDIM, DDIM, 1.f, 0, 0);
    ln_residual_kernel<<<(unsigned)MQ, 256, 0, stream>>>(tgt, t2,
        ln2_g + (size_t)i * DDIM, ln2_b + (size_t)i * DDIM);

    // ---- attention box bounds ----
    const float p0 = FOCUS[i] * 20.f, p1 = FOCUS[i] * 20.f, p2 = FOCUS[i] * 32.f;
    if (i == 0) {
      bounds_kernel<<<(unsigned)((MQ + 127) / 128), 128, 0, stream>>>(
          h2, reg_w3, reg_b3, anchors, attnarea, bnds, 0, p0, p1, p2);
    } else {
      gemm_xwT<<<gQ, 128, 0, stream>>>(tgt, reg_w1, reg_b1, h1, (int)MQ, DDIM, DDIM, 1.f, 1, 0);
      gemm_xwT<<<gQ, 128, 0, stream>>>(h1, reg_w2, reg_b2, h2, (int)MQ, DDIM, DDIM, 1.f, 1, 0);
      bounds_kernel<<<(unsigned)((MQ + 127) / 128), 128, 0, stream>>>(
          h2, reg_w3, reg_b3, anchors, attnarea, bnds, 1, p0, p1, p2);
    }

    // ---- cross attention ----
    add_qpos_kernel<<<gEl, 256, 0, stream>>>(tgt, qe, qk);
    gemm_xwT<<<gQ, 128, 0, stream>>>(qk, ca_k_w + (size_t)i * DDIM * DDIM, nullptr, qc,
                                     (int)MQ, DDIM, DDIM, qscale, 0, 1);
    gemm_xwT<<<gP, 128, 0, stream>>>(kvT, ca_k_w + (size_t)i * DDIM * DDIM, nullptr, Kc,
                                     (int)MP, DDIM, DDIM, 1.f, 0, 1);
    gemm_xwT<<<gP, 128, 0, stream>>>(srcT, ca_v_w + (size_t)i * DDIM * DDIM, nullptr, Vc,
                                     (int)MP, DDIM, DDIM, 1.f, 0, 1);
    flash_attn<<<gFlash, 128, 0, stream>>>(qc, Kc, Vc, ob, bnds, NQ, PTOT, 1);
    gemm_xwT<<<gQ, 128, 0, stream>>>(ob, ca_p_w + (size_t)i * DDIM * DDIM,
                                     ca_p_b + (size_t)i * DDIM, t2,
                                     (int)MQ, DDIM, DDIM, 1.f, 0, 0);
    ln_residual_kernel<<<(unsigned)MQ, 256, 0, stream>>>(tgt, t2,
        ln1_g + (size_t)i * DDIM, ln1_b + (size_t)i * DDIM);

    // ---- FFN ----
    gemm_xwT<<<gF, 128, 0, stream>>>(tgt, ffn_w1 + (size_t)i * FF * DDIM,
                                     ffn_b1 + (size_t)i * FF, h1, (int)MQ, FF, DDIM, 1.f, 1, 0);
    gemm_xwT<<<gQ, 128, 0, stream>>>(h1, ffn_w2 + (size_t)i * DDIM * FF,
                                     ffn_b2 + (size_t)i * DDIM, t2, (int)MQ, DDIM, FF, 1.f, 0, 0);
    ln_residual_kernel<<<(unsigned)MQ, 256, 0, stream>>>(tgt, t2,
        ln3_g + (size_t)i * DDIM, ln3_b + (size_t)i * DDIM);
  }

  copy_kernel<<<(unsigned)((out_size + 255) / 256), 256, 0, stream>>>(tgt, (float*)d_out, out_size);
}
